// EEGWordClsModel_22720376996340
// MI455X (gfx1250) — matrix operand order repaired and checked by reading, not compiled, not run on hardware
//
#include <hip/hip_runtime.h>
#include <hip/hip_bf16.h>
#include <math.h>

typedef __attribute__((ext_vector_type(16))) _Float16 v16h;
typedef __attribute__((ext_vector_type(8)))  float    v8f;
typedef __attribute__((ext_vector_type(2)))  _Float16 v2h;
typedef __attribute__((ext_vector_type(4)))  _Float16 v4h;

// ---------------- tiling ----------------
static constexpr int BM = 128, BN = 64, BKT = 64;
static constexpr int SAP = BKT + 8;   // padded LDS row pitch (halves): 144B -> bank-conflict free

// ---------------- problem dimensions ----------------
static constexpr int Bc    = 64;    // batch
static constexpr int T0c   = 396;
static constexpr int FINc  = 128;
static constexpr int DMc   = 768;
static constexpr int NHc   = 8;
static constexpr int DAc   = 96;
static constexpr int DFFc  = 3072;
static constexpr int NLc   = 6;
static constexpr int NOUTc = 256;
static constexpr int RELDc = 100;
static constexpr int T1c   = 198;   // after rb1 (stride 2)
static constexpr int T1p   = 200;   // padded
static constexpr int T2c   = 99;    // after rb2 (stride 2)
static constexpr int T2p   = 100;   // padded
static constexpr int Lc    = 100;   // seq len with cls
static constexpr int PWc   = 199;   // 2L-1

// ---------------- helpers ----------------
__device__ __forceinline__ v4h pack4(float x, float y, float z, float w) {
  v2h lo = __builtin_bit_cast(v2h, __builtin_amdgcn_cvt_pkrtz(x, y));
  v2h hi = __builtin_bit_cast(v2h, __builtin_amdgcn_cvt_pkrtz(z, w));
  v4h r; r[0] = lo[0]; r[1] = lo[1]; r[2] = hi[0]; r[3] = hi[1];
  return r;
}

// WMMA fragment loaders (ISA 7.12.2 layouts)
__device__ __forceinline__ v16h frag_a(const _Float16* s, int row0, int kb, int lane) {
  // A 16x32 f16: lane<16 row M=lane holds K {0..7,16..23}; lane>=16 holds K {8..15,24..31}
  int m = lane & 15, hi = lane >> 4;
  const _Float16* p = s + (row0 + m) * SAP + kb + hi * 8;
  v16h f;
#pragma unroll
  for (int j = 0; j < 8; ++j) f[j] = p[j];
#pragma unroll
  for (int j = 0; j < 8; ++j) f[j + 8] = p[j + 16];
  return f;
}

__device__ __forceinline__ v16h frag_b(const _Float16* s, int col0, int kb, int lane) {
  // B tile in LDS as [n][k]: lane<16 col N=lane K {0..7,16..23}; lane>=16 K {8..15,24..31}
  int n = lane & 15, hi = lane >> 4;
  const _Float16* p = s + (col0 + n) * SAP + kb + hi * 8;
  v16h f;
#pragma unroll
  for (int j = 0; j < 8; ++j) f[j] = p[j];
#pragma unroll
  for (int j = 0; j < 8; ++j) f[j + 8] = p[j + 16];
  return f;
}

// ---------------- generalized batched WMMA GEMM ----------------
// C[z] = act( (A[z]*B[z])*alpha (opt *scaleM+shiftM) (opt +biasN) (opt +resid[z]) )
// z-operand addressing: base + (z/div)*so + (z%div)*si.
__global__ __launch_bounds__(256) void wmma_gemm_kernel(
    const float* __restrict__ A, long long a_so, int a_div, long long a_si, int lda,
    const float* __restrict__ Bm, long long b_so, int b_div, long long b_si, int ldb, int btr,
    float* __restrict__ C, long long c_so, int c_div, long long c_si, int ldc,
    const float* __restrict__ resid,
    int M, int N, int K, float alpha,
    const float* __restrict__ biasN,
    const float* __restrict__ scaleM, const float* __restrict__ shiftM,
    int relu)
{
  __shared__ _Float16 As[BM * SAP];
  __shared__ _Float16 Bs[BN * SAP];
  const int tid  = threadIdx.x;
  const int lane = tid & 31;
  const int wid  = tid >> 5;
  const int wm   = (wid & 3) << 5;   // wave M offset: 0,32,64,96
  const int wn   = (wid >> 2) << 5;  // wave N offset: 0,32
  const long long z = blockIdx.z;
  const float* pA = A  + (z / a_div) * a_so + (z % a_div) * a_si;
  const float* pB = Bm + (z / b_div) * b_so + (z % b_div) * b_si;
  float*       pC = C  + (z / c_div) * c_so + (z % c_div) * c_si;
  const float* pR = resid ? (resid + (z / c_div) * c_so + (z % c_div) * c_si) : nullptr;
  const int m0 = blockIdx.y * BM;
  const int n0 = blockIdx.x * BN;
  const bool mFull = (m0 + BM <= M);
  const bool nFull = (n0 + BN <= N);
  v8f acc[2][2] = {};

  for (int k0 = 0; k0 < K; k0 += BKT) {
    const bool kFull = (k0 + BKT <= K);
    // ---- stage A tile ----
    if (mFull && kFull) {            // interior: unguarded, all loads in flight
      float4 ta[8];
#pragma unroll
      for (int u = 0; u < 8; ++u) {
        int i = tid + u * 256;
        int r = i >> 4, c = (i & 15) << 2;
        ta[u] = *(const float4*)(pA + (long long)(m0 + r) * lda + (k0 + c));
      }
#pragma unroll
      for (int u = 0; u < 8; ++u) {
        int i = tid + u * 256;
        int r = i >> 4, c = (i & 15) << 2;
        *(v4h*)(As + r * SAP + c) = pack4(ta[u].x, ta[u].y, ta[u].z, ta[u].w);
      }
    } else {
      for (int i = tid; i < (BM * BKT) / 4; i += 256) {
        int r = i >> 4, c = (i & 15) << 2;
        int gm = m0 + r, gk = k0 + c;
        float e0 = (gm < M && gk + 0 < K) ? pA[(long long)gm * lda + gk + 0] : 0.f;
        float e1 = (gm < M && gk + 1 < K) ? pA[(long long)gm * lda + gk + 1] : 0.f;
        float e2 = (gm < M && gk + 2 < K) ? pA[(long long)gm * lda + gk + 2] : 0.f;
        float e3 = (gm < M && gk + 3 < K) ? pA[(long long)gm * lda + gk + 3] : 0.f;
        *(v4h*)(As + r * SAP + c) = pack4(e0, e1, e2, e3);
      }
    }
    // ---- stage B tile into [n][k] ----
    if (btr) {                       // B is (N,K) row-major: contiguous along k
      if (nFull && kFull) {
        float4 tb[4];
#pragma unroll
        for (int u = 0; u < 4; ++u) {
          int i = tid + u * 256;
          int r = i >> 4, c = (i & 15) << 2;
          tb[u] = *(const float4*)(pB + (long long)(n0 + r) * ldb + (k0 + c));
        }
#pragma unroll
        for (int u = 0; u < 4; ++u) {
          int i = tid + u * 256;
          int r = i >> 4, c = (i & 15) << 2;
          *(v4h*)(Bs + r * SAP + c) = pack4(tb[u].x, tb[u].y, tb[u].z, tb[u].w);
        }
      } else {
        for (int i = tid; i < (BN * BKT) / 4; i += 256) {
          int r = i >> 4, c = (i & 15) << 2;
          int gn = n0 + r, gk = k0 + c;
          float e0 = (gn < N && gk + 0 < K) ? pB[(long long)gn * ldb + gk + 0] : 0.f;
          float e1 = (gn < N && gk + 1 < K) ? pB[(long long)gn * ldb + gk + 1] : 0.f;
          float e2 = (gn < N && gk + 2 < K) ? pB[(long long)gn * ldb + gk + 2] : 0.f;
          float e3 = (gn < N && gk + 3 < K) ? pB[(long long)gn * ldb + gk + 3] : 0.f;
          *(v4h*)(Bs + r * SAP + c) = pack4(e0, e1, e2, e3);
        }
      }
    } else {                         // B is (K,N) row-major: 4x4 register transpose
      int k4 = (tid & 15) << 2;
      int n4 = (tid >> 4) << 2;      // 256 threads cover the 16x16 grid of 4x4 micro-tiles
      int gk = k0 + k4, gn = n0 + n4;
      float rv[4][4];
      if (nFull && kFull) {
        const float* bp = pB + (long long)gk * ldb + gn;
#pragma unroll
        for (int j = 0; j < 4; ++j) {
          const float4 v = *(const float4*)(bp + (long long)j * ldb);
          rv[j][0] = v.x; rv[j][1] = v.y; rv[j][2] = v.z; rv[j][3] = v.w;
        }
      } else {
#pragma unroll
        for (int j = 0; j < 4; ++j)
#pragma unroll
          for (int n = 0; n < 4; ++n)
            rv[j][n] = (gk + j < K && gn + n < N) ? pB[(long long)(gk + j) * ldb + gn + n] : 0.f;
      }
#pragma unroll
      for (int n = 0; n < 4; ++n)
        *(v4h*)(Bs + (n4 + n) * SAP + k4) = pack4(rv[0][n], rv[1][n], rv[2][n], rv[3][n]);
    }
    // speculative prefetch of next K tile (global_prefetch_b8)
    if (k0 + BKT < K) {
      __builtin_prefetch(pA + (long long)(m0 + (tid & (BM - 1))) * lda + (k0 + BKT), 0, 3);
      if (btr)
        __builtin_prefetch(pB + (long long)(n0 + (tid & (BN - 1))) * ldb + (k0 + BKT), 0, 3);
      else
        __builtin_prefetch(pB + (long long)(k0 + BKT + (tid & 63)) * ldb + n0, 0, 3);
    }
    __syncthreads();
#pragma unroll
    for (int kk = 0; kk < BKT; kk += 32) {
      v16h a0 = frag_a(As, wm,      kk, lane);
      v16h a1 = frag_a(As, wm + 16, kk, lane);
      v16h b0 = frag_b(Bs, wn,      kk, lane);
      v16h b1 = frag_b(Bs, wn + 16, kk, lane);
      acc[0][0] = __builtin_amdgcn_wmma_f32_16x16x32_f16(false, a0, false, b0, (short)0, acc[0][0], false, false);
      acc[0][1] = __builtin_amdgcn_wmma_f32_16x16x32_f16(false, a0, false, b1, (short)0, acc[0][1], false, false);
      acc[1][0] = __builtin_amdgcn_wmma_f32_16x16x32_f16(false, a1, false, b0, (short)0, acc[1][0], false, false);
      acc[1][1] = __builtin_amdgcn_wmma_f32_16x16x32_f16(false, a1, false, b1, (short)0, acc[1][1], false, false);
    }
    __syncthreads();
  }

  // epilogue; C frag layout: reg r -> (M = r + (lane>>4)*8, N = lane&15)
  const int ln = lane & 15, lh = lane >> 4;
#pragma unroll
  for (int am = 0; am < 2; ++am) {
#pragma unroll
    for (int r = 0; r < 8; ++r) {
      int gm = m0 + wm + am * 16 + r + lh * 8;
      if (gm >= M) continue;
      float sc = 1.f, sh = 0.f;
      if (scaleM) { sc = scaleM[gm]; sh = shiftM[gm]; }
#pragma unroll
      for (int bn = 0; bn < 2; ++bn) {
        int gn = n0 + wn + bn * 16 + ln;
        if (gn >= N) continue;
        float v = acc[am][bn][r] * alpha;
        if (scaleM) v = v * sc + sh;
        if (biasN)  v += biasN[gn];
        if (pR)     v += pR[(long long)gm * ldc + gn];
        if (relu)   v = fmaxf(v, 0.f);
        pC[(long long)gm * ldc + gn] = v;
      }
    }
  }
}

// ---------------- helper kernels ----------------
// in (B, Xd, Yd) with row pitch ldin -> out (B, Yd, Xd) dense
__global__ void transpose_xy_kernel(const float* __restrict__ in, float* __restrict__ out,
                                    int Xd, int Yd, int ldin, long long total) {
  long long idx = (long long)blockIdx.x * blockDim.x + threadIdx.x;
  if (idx >= total) return;
  int y = (int)(idx % Yd); long long u = idx / Yd;
  int xx = (int)(u % Xd); int b = (int)(u / Xd);
  out[((long long)b * Yd + y) * Xd + xx] = in[((long long)b * Xd + xx) * ldin + y];
}

// x (B,C,Tin) pitch ldx -> col (B, C*KW, Tout) pitch ldcol, row index c*KW+k
__global__ void im2col_kernel(const float* __restrict__ x, float* __restrict__ col,
                              int C, int Tin, int ldx, int Tout, int ldcol,
                              int KW, int stride, int pad, long long total) {
  long long idx = (long long)blockIdx.x * blockDim.x + threadIdx.x;
  if (idx >= total) return;
  int t = (int)(idx % Tout); long long u = idx / Tout;
  int k = (int)(u % KW); u /= KW;
  int c = (int)(u % C);  int b = (int)(u / C);
  int ti = t * stride + k - pad;
  float v = 0.f;
  if (ti >= 0 && ti < Tin) v = x[((long long)b * C + c) * ldx + ti];
  col[((long long)b * C * KW + (long long)c * KW + k) * ldcol + t] = v;
}

__global__ void fuse_bn_kernel(float* __restrict__ sc, float* __restrict__ sh,
                               const float* __restrict__ cb, const float* __restrict__ bs,
                               const float* __restrict__ bb, int n) {
  int i = blockIdx.x * blockDim.x + threadIdx.x;
  if (i < n) { sc[i] = bs[i]; sh[i] = cb[i] * bs[i] + bb[i]; }
}

__global__ void set_cls_kernel(float* __restrict__ X, const float* __restrict__ cls, int n) {
  int idx = blockIdx.x * blockDim.x + threadIdx.x;
  if (idx >= n) return;
  int b = idx / DMc, c = idx % DMc;
  X[(long long)b * Lc * DMc + c] = cls[c];
}

// softmax over j of logits[z,i,j] + pos[z,i, j-i+L-1]; layouts (HB,L,L) and (HB,L,PW)
__global__ __launch_bounds__(128) void softmax_rel_kernel(const float* __restrict__ logits,
                                                          const float* __restrict__ pos,
                                                          float* __restrict__ probs) {
  int row = blockIdx.x;
  int z = row / Lc, i = row - z * Lc;
  const float* lrow = logits + (long long)z * Lc * Lc + (long long)i * Lc;
  const float* prow = pos + (long long)z * Lc * PWc + (long long)i * PWc + (Lc - 1 - i);
  float* orow = probs + (long long)z * Lc * Lc + (long long)i * Lc;
  int j = threadIdx.x;
  float v = -3.0e38f;
  if (j < Lc) v = lrow[j] + prow[j];
  __shared__ float red[128];
  red[j] = v; __syncthreads();
  for (int s = 64; s > 0; s >>= 1) { if (j < s) red[j] = fmaxf(red[j], red[j + s]); __syncthreads(); }
  float mx = red[0]; __syncthreads();
  float e = (j < Lc) ? __expf(v - mx) : 0.f;
  red[j] = e; __syncthreads();
  for (int s = 64; s > 0; s >>= 1) { if (j < s) red[j] += red[j + s]; __syncthreads(); }
  float inv = 1.f / red[0];
  if (j < Lc) orow[j] = e * inv;
}

// x[row] = LayerNorm(x[row] + a[row]) * g + b
__global__ __launch_bounds__(256) void add_ln_kernel(float* __restrict__ x, const float* __restrict__ a,
                                                     const float* __restrict__ g, const float* __restrict__ b) {
  int row = blockIdx.x, tid = threadIdx.x;
  float* xr = x + (long long)row * DMc;
  const float* ar = a + (long long)row * DMc;
  float loc[3]; float s = 0.f;
#pragma unroll
  for (int e = 0; e < 3; ++e) { int c = tid + e * 256; float v = xr[c] + ar[c]; loc[e] = v; s += v; }
  __shared__ float red[256];
  red[tid] = s; __syncthreads();
  for (int st = 128; st > 0; st >>= 1) { if (tid < st) red[tid] += red[tid + st]; __syncthreads(); }
  float mean = red[0] * (1.f / DMc); __syncthreads();
  float vs = 0.f;
#pragma unroll
  for (int e = 0; e < 3; ++e) { float d = loc[e] - mean; vs += d * d; }
  red[tid] = vs; __syncthreads();
  for (int st = 128; st > 0; st >>= 1) { if (tid < st) red[tid] += red[tid + st]; __syncthreads(); }
  float rstd = rsqrtf(red[0] * (1.f / DMc) + 1e-5f);
#pragma unroll
  for (int e = 0; e < 3; ++e) { int c = tid + e * 256; xr[c] = (loc[e] - mean) * rstd * g[c] + b[c]; }
}

// ---------------- host side ----------------
static void launch_gemm(hipStream_t s,
    const float* A, long long a_so, int a_div, long long a_si, int lda,
    const float* B, long long b_so, int b_div, long long b_si, int ldb, int btr,
    float* C, long long c_so, int c_div, long long c_si, int ldc,
    const float* resid, int M, int N, int K, float alpha,
    const float* biasN, const float* scaleM, const float* shiftM, int relu, int batch)
{
  dim3 g((N + BN - 1) / BN, (M + BM - 1) / BM, batch);
  wmma_gemm_kernel<<<g, 256, 0, s>>>(A, a_so, a_div, a_si, lda,
                                     B, b_so, b_div, b_si, ldb, btr,
                                     C, c_so, c_div, c_si, ldc,
                                     resid, M, N, K, alpha, biasN, scaleM, shiftM, relu);
}

// workspace layout (bytes)
static constexpr size_t OFF_BIG = 0;                                // 132 MB: im2col / QKV / FFN hidden
static constexpr size_t OFF_H   = OFF_BIG + (132ull << 20);         // 40 MB: conv h / logits / o
static constexpr size_t OFF_R   = OFF_H   + (40ull  << 20);         // 40 MB: conv h2,x2 / probs
static constexpr size_t OFF_POS = OFF_R   + (40ull  << 20);         // 44 MB: xT / pos / attn / ffn_out
static constexpr size_t OFF_X   = OFF_POS + (44ull  << 20);         // 20 MB: main sequence x (B,100,768)
static constexpr size_t OFF_SS  = OFF_X   + (20ull  << 20);         // 64 KB: fused bn coefficients

static void run_resblock(hipStream_t s, const float* xin, int Cin, int Tin, int ldx,
                         int Tout, int Tout_p, int stride, const float* const* p,
                         float* col, float* hbuf, float* h2buf, float* xout, float* ss)
{
  const float *c1w = p[0], *c1b = p[1], *s1 = p[2], *b1 = p[3];
  const float *c2w = p[4], *c2b = p[5], *s2 = p[6], *b2 = p[7];
  const float *crw = p[8], *crb = p[9], *sr = p[10], *br = p[11];
  float *sc1 = ss,           *sh1 = ss + DMc;
  float *sc2 = ss + 2 * DMc, *sh2 = ss + 3 * DMc;
  float *scr = ss + 4 * DMc, *shr = ss + 5 * DMc;
  fuse_bn_kernel<<<3, 256, 0, s>>>(sc1, sh1, c1b, s1, b1, DMc);
  fuse_bn_kernel<<<3, 256, 0, s>>>(sc2, sh2, c2b, s2, b2, DMc);
  fuse_bn_kernel<<<3, 256, 0, s>>>(scr, shr, crb, sr, br, DMc);

  // conv1 (k=3, stride) + bn1 + relu -> hbuf
  long long t1 = (long long)Bc * Cin * 3 * Tout;
  im2col_kernel<<<(unsigned)((t1 + 255) / 256), 256, 0, s>>>(xin, col, Cin, Tin, ldx, Tout, Tout_p, 3, stride, 1, t1);
  launch_gemm(s, c1w, 0, 1, 0, Cin * 3,
              col, (long long)Cin * 3 * Tout_p, 1, 0, Tout_p, 0,
              hbuf, (long long)DMc * Tout_p, 1, 0, Tout_p,
              nullptr, DMc, Tout, Cin * 3, 1.f, nullptr, sc1, sh1, 1, Bc);
  // conv2 (k=3, stride 1) + bn2 -> h2buf
  long long t2 = (long long)Bc * DMc * 3 * Tout;
  im2col_kernel<<<(unsigned)((t2 + 255) / 256), 256, 0, s>>>(hbuf, col, DMc, Tout, Tout_p, Tout, Tout_p, 3, 1, 1, t2);
  launch_gemm(s, c2w, 0, 1, 0, DMc * 3,
              col, (long long)DMc * 3 * Tout_p, 1, 0, Tout_p, 0,
              h2buf, (long long)DMc * Tout_p, 1, 0, Tout_p,
              nullptr, DMc, Tout, DMc * 3, 1.f, nullptr, sc2, sh2, 0, Bc);
  // shortcut (k=1, stride) + bnr + h2 + relu -> xout
  long long tr = (long long)Bc * Cin * Tout;
  im2col_kernel<<<(unsigned)((tr + 255) / 256), 256, 0, s>>>(xin, col, Cin, Tin, ldx, Tout, Tout_p, 1, stride, 0, tr);
  launch_gemm(s, crw, 0, 1, 0, Cin,
              col, (long long)Cin * Tout_p, 1, 0, Tout_p, 0,
              xout, (long long)DMc * Tout_p, 1, 0, Tout_p,
              h2buf, DMc, Tout, Cin, 1.f, nullptr, scr, shr, 1, Bc);
}

extern "C" void kernel_launch(void* const* d_in, const int* in_sizes, int n_in,
                              void* d_out, int out_size, void* d_ws, size_t ws_size,
                              hipStream_t stream) {
  (void)in_sizes; (void)n_in; (void)out_size; (void)ws_size;
  auto in = [&](int i) { return (const float*)d_in[i]; };
  char* ws = (char*)d_ws;
  auto wf = [&](size_t off) { return (float*)(ws + off); };

  float* xT   = wf(OFF_POS);   // (B,128,396), dead before pos_all is first used
  float* colB = wf(OFF_BIG);
  float* hH   = wf(OFF_H);
  float* hR   = wf(OFF_R);
  float* ss   = wf(OFF_SS);
  float* X    = wf(OFF_X);     // (B, 100, 768) persistent

  // --- stage 0: (B,T,F) -> (B,F,T) ---
  long long tt0 = (long long)Bc * T0c * FINc;
  transpose_xy_kernel<<<(unsigned)((tt0 + 255) / 256), 256, 0, stream>>>(in(0), xT, T0c, FINc, FINc, tt0);

  // --- res blocks (conv outputs use padded time pitch for aligned vector loads) ---
  const float* rb1p[12] = { in(1), in(2), in(3), in(4), in(5), in(6), in(7), in(8), in(9), in(10), in(11), in(12) };
  run_resblock(stream, xT, FINc, T0c, T0c, T1c, T1p, 2, rb1p, colB, hH, hR, hH, ss);          // x1 -> hH (pitch 200)
  const float* rb2p[12] = { in(13), in(14), in(15), in(16), in(17), in(18), in(19), in(20), in(21), in(22), in(23), in(24) };
  run_resblock(stream, hH, DMc, T1c, T1p, T2c, T2p, 2, rb2p, colB, hR, wf(OFF_POS), hR, ss);  // x2 -> hR (pitch 100)

  // --- (B,768,99 pitch 100) -> (B*99,768), then w_in + bias into X rows 1..99; cls row 0 ---
  float* xt2 = wf(OFF_BIG);
  long long tt1 = (long long)Bc * DMc * T2c;
  transpose_xy_kernel<<<(unsigned)((tt1 + 255) / 256), 256, 0, stream>>>(hR, xt2, DMc, T2c, T2p, tt1);
  launch_gemm(stream, xt2, (long long)T2c * DMc, 1, 0, DMc,
              in(25), 0, 1, 0, DMc, 1,
              X + DMc, (long long)Lc * DMc, 1, 0, DMc,
              nullptr, T2c, DMc, DMc, 1.f, in(26), nullptr, nullptr, 0, Bc);
  set_cls_kernel<<<(Bc * DMc + 255) / 256, 256, 0, stream>>>(X, in(27), Bc * DMc);

  // --- transformer layers ---
  float* Q      = wf(OFF_BIG);
  float* Kb     = wf(OFF_BIG + (24ull << 20));
  float* V      = wf(OFF_BIG + (48ull << 20));
  float* logits = wf(OFF_H);
  float* probs  = wf(OFF_R);
  float* pos    = wf(OFF_POS);
  float* obuf   = wf(OFF_H);
  float* attn   = wf(OFF_POS);
  float* ffh    = wf(OFF_BIG);
  float* ffo    = wf(OFF_POS + (22ull << 20));
  const int HB = NHc * Bc;                      // 512 batched GEMM slices, z = h*B + b
  const float qscale = 1.0f / sqrtf((float)DAc);

  for (int l = 0; l < NLc; ++l) {
    const float* wq  = in(28) + (size_t)l * NHc * DMc * DAc;
    const float* wk  = in(29) + (size_t)l * NHc * DMc * DAc;
    const float* wv  = in(30) + (size_t)l * NHc * DMc * DAc;
    const float* wo  = in(31) + (size_t)l * NHc * DAc * DMc;
    const float* rel = in(32) + (size_t)l * NHc * (2 * RELDc + 1) * DAc; // start = RELD-L = 0

    // Q/K/V: (H,B,L,A); A = x[b] (z%B), B = w[h] (z/B)
    launch_gemm(stream, X, 0, Bc, (long long)Lc * DMc, DMc,
                wq, (long long)DMc * DAc, Bc, 0, DAc, 0,
                Q, (long long)Lc * DAc, 1, 0, DAc,
                nullptr, Lc, DAc, DMc, qscale, nullptr, nullptr, nullptr, 0, HB);
    launch_gemm(stream, X, 0, Bc, (long long)Lc * DMc, DMc,
                wk, (long long)DMc * DAc, Bc, 0, DAc, 0,
                Kb, (long long)Lc * DAc, 1, 0, DAc,
                nullptr, Lc, DAc, DMc, 1.f, nullptr, nullptr, nullptr, 0, HB);
    launch_gemm(stream, X, 0, Bc, (long long)Lc * DMc, DMc,
                wv, (long long)DMc * DAc, Bc, 0, DAc, 0,
                V, (long long)Lc * DAc, 1, 0, DAc,
                nullptr, Lc, DAc, DMc, 1.f, nullptr, nullptr, nullptr, 0, HB);
    // logits = Q K^T  (HB, L, L)
    launch_gemm(stream, Q, (long long)Lc * DAc, 1, 0, DAc,
                Kb, (long long)Lc * DAc, 1, 0, DAc, 1,
                logits, (long long)Lc * Lc, 1, 0, Lc,
                nullptr, Lc, Lc, DAc, 1.f, nullptr, nullptr, nullptr, 0, HB);
    // pos = Q rel^T  (HB, L, 199); rel per head (z/B), row stride within 201-row table
    launch_gemm(stream, Q, (long long)Lc * DAc, 1, 0, DAc,
                rel, (long long)(2 * RELDc + 1) * DAc, Bc, 0, DAc, 1,
                pos, (long long)Lc * PWc, 1, 0, PWc,
                nullptr, Lc, PWc, DAc, 1.f, nullptr, nullptr, nullptr, 0, HB);
    // softmax(logits + gathered pos)
    softmax_rel_kernel<<<HB * Lc, 128, 0, stream>>>(logits, pos, probs);
    // O = P V, written into (B, L, H*A): h = z/B -> +h*96, b = z%B -> +b*L*768
    launch_gemm(stream, probs, (long long)Lc * Lc, 1, 0, Lc,
                V, (long long)Lc * DAc, 1, 0, DAc, 0,
                obuf, (long long)DAc, Bc, (long long)Lc * DMc, DMc,
                nullptr, Lc, DAc, Lc, 1.f, nullptr, nullptr, nullptr, 0, HB);
    // attn = O_cat (6400,768) x wo_flat (768,768)
    launch_gemm(stream, obuf, 0, 1, 0, DMc,
                wo, 0, 1, 0, DMc, 0,
                attn, 0, 1, 0, DMc,
                nullptr, Bc * Lc, DMc, DMc, 1.f, nullptr, nullptr, nullptr, 0, 1);
    // x = LN(x + attn)
    add_ln_kernel<<<Bc * Lc, 256, 0, stream>>>(X, attn, in(33) + (size_t)l * DMc, in(34) + (size_t)l * DMc);
    // FFN
    launch_gemm(stream, X, 0, 1, 0, DMc,
                in(37) + (size_t)l * DFFc * DMc, 0, 1, 0, DMc, 1,
                ffh, 0, 1, 0, DFFc,
                nullptr, Bc * Lc, DFFc, DMc, 1.f, in(38) + (size_t)l * DFFc, nullptr, nullptr, 1, 1);
    launch_gemm(stream, ffh, 0, 1, 0, DFFc,
                in(39) + (size_t)l * DMc * DFFc, 0, 1, 0, DFFc, 1,
                ffo, 0, 1, 0, DMc,
                nullptr, Bc * Lc, DMc, DFFc, 1.f, in(40) + (size_t)l * DMc, nullptr, nullptr, 0, 1);
    // x = LN(x + ffn)
    add_ln_kernel<<<Bc * Lc, 256, 0, stream>>>(X, ffo, in(35) + (size_t)l * DMc, in(36) + (size_t)l * DMc);
  }

  // --- classifier: out = x[:,0,:] @ w_out.T + b ; cls rows have stride L*768 ---
  launch_gemm(stream, X, 0, 1, 0, Lc * DMc,
              in(41), 0, 1, 0, DMc, 1,
              (float*)d_out, 0, 1, 0, NOUTc,
              nullptr, Bc, NOUTc, DMc, 1.f, in(42), nullptr, nullptr, 0, 1);
}
